// ModelTorch_33002528702615
// MI455X (gfx1250) — compile-verified, matrix operand order checked by reading, not run on hardware
//
#include <hip/hip_runtime.h>

// ---------------------------------------------------------------------------
// Problem constants (from reference setup_inputs)
// ---------------------------------------------------------------------------
#define DIM      128
#define NFEAT    2048
#define NTRAIN   50000
#define NCHUNKS  1563                 // ceil(50000 / 32)
#define NPAD     (NCHUNKS * 32)       // 50016, zero-padded tail
#define OUTD     64
#define SPLIT    16                   // N-dimension split-K factor
#define CPS      ((NCHUNKS + SPLIT - 1) / SPLIT)   // chunks per split = 98

// k = exp(-dist / RBF_CO / 2)  ->  exp(dist * KEXP)
#define KEXP (-1.0f / (2.0f * 10.077141124806595f))

typedef __bf16 bf16_t;
typedef __attribute__((ext_vector_type(16))) __bf16 bv16;
typedef __attribute__((ext_vector_type(8)))  __bf16 bv8;
typedef __attribute__((ext_vector_type(4)))  __bf16 bv4;
typedef __attribute__((ext_vector_type(8)))  float  v8f;
typedef __attribute__((ext_vector_type(4)))  float  v4f;

__device__ __forceinline__ v8f wmma_bf16(bv16 a, bv16 b, v8f c) {
    // (neg_a, A, neg_b, B, c_mod, C, reuse_a, reuse_b)
    return __builtin_amdgcn_wmma_f32_16x16x32_bf16(false, a, false, b,
                                                   (short)0, c, false, false);
}

// ---------------------------------------------------------------------------
// Zero the output (harness poisons d_out; we accumulate with atomics)
// ---------------------------------------------------------------------------
__global__ __launch_bounds__(256) void zero_out(float* __restrict__ out, int n) {
    int i = blockIdx.x * 256 + threadIdx.x;
    if (i < n) out[i] = 0.0f;
}

// ---------------------------------------------------------------------------
// Pre-split train_data (f32) into bf16 hi/lo arrays laid out exactly as the
// WMMA B-operand expects, and precompute t2[j] = |t_j|^2.
//
// B-operand (16-bit, 32x16) lane layout: lane l holds column N = l%16 with
// h = l/16 selecting K-halves:  elem idx 0..7  -> K = 32c + 8h + i
//                               elem idx 8..15 -> K = 32c + 16 + 8h + (i-8)
// Storage:  Thi[row*128 + c*32 + h*16 + g*8 + i] = train[row][32c + 16g + 8h + i]
// so the main kernel reads 32 contiguous bytes per (col, c, h).
// 32 lanes cooperate per row; rows >= NTRAIN are zero-filled.
// ---------------------------------------------------------------------------
__global__ __launch_bounds__(256) void prep_train(const float* __restrict__ train,
                                                  bf16_t* __restrict__ Thi,
                                                  bf16_t* __restrict__ Tlo,
                                                  float* __restrict__ t2) {
    int row  = blockIdx.x * 8 + (threadIdx.x >> 5);
    int lane = threadIdx.x & 31;
    if (row >= NPAD) return;

    int k0 = lane * 4;                 // 4 consecutive K per lane
    v4f s = {0.0f, 0.0f, 0.0f, 0.0f};
    if (row < NTRAIN) s = *(const v4f*)(train + (size_t)row * DIM + k0);

    bv4 hv, lv;
    float p = 0.0f;
#pragma unroll
    for (int j = 0; j < 4; ++j) {
        float x = s[j];
        bf16_t hb = (bf16_t)x;
        hv[j] = hb;
        lv[j] = (bf16_t)(x - (float)hb);
        p += x * x;
    }
    int c = k0 >> 5, g = (k0 >> 4) & 1, hh = (k0 >> 3) & 1, i0 = k0 & 7;
    size_t dst = (size_t)row * DIM + c * 32 + hh * 16 + g * 8 + i0;
    *(bv4*)(Thi + dst) = hv;
    *(bv4*)(Tlo + dst) = lv;

#pragma unroll
    for (int off = 16; off >= 1; off >>= 1) p += __shfl_xor(p, off, 32);
    if (lane == 0) t2[row] = p;        // zero rows give p == 0
}

// ---------------------------------------------------------------------------
// Pre-swizzle weights (f32 -> bf16) into per-32-chunk B-operand order:
//   W2[jc*2048 + o*512 + (h*16 + r)*16 + g*8 + i] = w[32jc + 16g + 8h + i][16o + r]
// so in the main loop lane l reads 32 contiguous bytes at  jc*2048 + o*512 + l*16.
// ---------------------------------------------------------------------------
__global__ __launch_bounds__(256) void prep_weights(const float* __restrict__ w,
                                                    bf16_t* __restrict__ W2) {
    int idx = blockIdx.x * 256 + threadIdx.x;      // over NPAD * 16
    if (idx >= NPAD * 16) return;
    int row = idx >> 4, q = idx & 15;              // cols 4q .. 4q+3
    v4f s = {0.0f, 0.0f, 0.0f, 0.0f};
    if (row < NTRAIN) s = *(const v4f*)(w + (size_t)row * OUTD + q * 4);

    int jc = row >> 5, K = row & 31;
    int g = K >> 4, hh = (K >> 3) & 1, i = K & 7;
    size_t base = (size_t)jc * 2048 + (size_t)(hh * 16) * 16 + g * 8 + i;
#pragma unroll
    for (int j = 0; j < 4; ++j) {
        int col = q * 4 + j;
        int o = col >> 4, r = col & 15;
        W2[base + (size_t)o * 512 + (size_t)r * 16] = (bf16_t)s[j];
    }
}

// ---------------------------------------------------------------------------
// Fused main kernel: one wave (32 threads) per block.
//   blockIdx.x : feature row-tile (16 rows)        [128 tiles]
//   blockIdx.y : split over train points           [SPLIT splits]
// Per 32-train-point chunk:
//   GEMM1: 2 x (4 K-chunks x 3 split-precision wmma) = 24 v_wmma_f32_16x16x32_bf16
//   elementwise: d2 -> sqrt -> exp -> bf16, staged through 1KB LDS (transpose)
//   GEMM2: 4 x wmma against pre-swizzled weights
// Output: f32 global atomic adds into [2048 x 64].
// ---------------------------------------------------------------------------
__global__ __launch_bounds__(32) void rbf_fused(const float* __restrict__ feats,
                                                const bf16_t* __restrict__ Thi,
                                                const bf16_t* __restrict__ Tlo,
                                                const float* __restrict__ t2,
                                                const bf16_t* __restrict__ W2,
                                                float* __restrict__ out) {
    const int tile  = blockIdx.x;
    const int split = blockIdx.y;
    const int lane  = threadIdx.x;
    const int r = lane & 15;
    const int h = lane >> 4;

    __shared__ float  x2s[16];
    __shared__ bf16_t klds[16 * 32];

    // ---- Load this wave's 16x128 feature tile into A-operand hi/lo regs ----
    // A-operand layout: lane l holds row M = l%16;  elem 0..7 -> K = 32c+8h+i,
    // elem 8..15 -> K = 32c+16+8h+(i-8).
    bv16 ahi[4], alo[4];
    float x2p = 0.0f;
    const float* frow = feats + (size_t)(tile * 16 + r) * DIM;
#pragma unroll
    for (int c = 0; c < 4; ++c) {
        float tmp[16];
        *(v4f*)(tmp + 0)  = *(const v4f*)(frow + c * 32 + h * 8);
        *(v4f*)(tmp + 4)  = *(const v4f*)(frow + c * 32 + h * 8 + 4);
        *(v4f*)(tmp + 8)  = *(const v4f*)(frow + c * 32 + 16 + h * 8);
        *(v4f*)(tmp + 12) = *(const v4f*)(frow + c * 32 + 16 + h * 8 + 4);
#pragma unroll
        for (int i = 0; i < 16; ++i) {
            float x = tmp[i];
            bf16_t hb = (bf16_t)x;
            ahi[c][i] = hb;
            alo[c][i] = (bf16_t)(x - (float)hb);
            x2p += x * x;
        }
    }
    // lanes l and l^16 hold complementary K-halves of the same row
    x2p += __shfl_xor(x2p, 16, 32);
    if (h == 0) x2s[r] = x2p;
    __syncthreads();
    float x2v[8];
#pragma unroll
    for (int v = 0; v < 8; ++v) x2v[v] = x2s[v + 8 * h];   // C-layout M = v + 8h

    v8f acco[4] = {};   // 16 rows x 64 out cols accumulator (C layout, 4 N-tiles)

    int jc0 = split * CPS;
    int jc1 = jc0 + CPS;
    if (jc1 > NCHUNKS) jc1 = NCHUNKS;

    for (int jc = jc0; jc < jc1; ++jc) {
        const int j0 = jc * 32;

        // ---------- GEMM1 + elementwise, two 16-column subtiles ----------
#pragma unroll
        for (int t = 0; t < 2; ++t) {
            const int col = j0 + t * 16 + r;
            const bf16_t* bh = Thi + (size_t)col * DIM + h * 16;
            const bf16_t* bl = Tlo + (size_t)col * DIM + h * 16;
            v8f a = {};
#pragma unroll
            for (int c = 0; c < 4; ++c) {
                bv16 bhi = *(const bv16*)(bh + c * 32);
                bv16 blo = *(const bv16*)(bl + c * 32);
                a = wmma_bf16(ahi[c], bhi, a);   // hi*hi
                a = wmma_bf16(ahi[c], blo, a);   // hi*lo
                a = wmma_bf16(alo[c], bhi, a);   // lo*hi  (~fp32 accurate dot)
            }
            const float tt = t2[col];
#pragma unroll
            for (int v = 0; v < 8; ++v) {
                float d2 = x2v[v] + tt - 2.0f * a[v];
                d2 = fmaxf(d2, 0.0f);
                float kk = __expf(sqrtf(d2) * KEXP);
                if (col >= NTRAIN) kk = 0.0f;    // exact tail handling
                klds[(v + 8 * h) * 32 + t * 16 + r] = (bf16_t)kk;
            }
        }
        __syncthreads();

        // ---------- transpose k-tile (C layout -> A layout) via LDS ----------
        bv16 a2;
        {
            bv8 g0 = *(const bv8*)(klds + r * 32 + 8 * h);
            bv8 g1 = *(const bv8*)(klds + r * 32 + 16 + 8 * h);
#pragma unroll
            for (int i = 0; i < 8; ++i) { a2[i] = g0[i]; a2[8 + i] = g1[i]; }
        }

        // ---------- GEMM2: k-tile @ weights-chunk (pre-swizzled) ----------
        const bf16_t* wc = W2 + (size_t)jc * 2048 + (size_t)lane * 16;
#pragma unroll
        for (int o = 0; o < 4; ++o) {
            bv16 b2 = *(const bv16*)(wc + (size_t)o * 512);
            acco[o] = wmma_bf16(a2, b2, acco[o]);
        }
        __syncthreads();   // klds reused next iteration
    }

    // ---------- accumulate split-K partial into global output ----------
#pragma unroll
    for (int o = 0; o < 4; ++o)
#pragma unroll
        for (int v = 0; v < 8; ++v)
            atomicAdd(out + (size_t)(tile * 16 + v + 8 * h) * OUTD + o * 16 + r,
                      acco[o][v]);
}

// ---------------------------------------------------------------------------
// Launch
// ---------------------------------------------------------------------------
extern "C" void kernel_launch(void* const* d_in, const int* in_sizes, int n_in,
                              void* d_out, int out_size, void* d_ws, size_t ws_size,
                              hipStream_t stream) {
    const float* feats = (const float*)d_in[0];   // [2048, 128]
    const float* train = (const float*)d_in[1];   // [50000, 128]
    const float* w     = (const float*)d_in[2];   // [50000, 64]
    float* out = (float*)d_out;                   // [2048, 64]

    // Workspace layout (bytes):
    //   Thi : NPAD*128*2 = 12,804,096
    //   Tlo : NPAD*128*2 = 12,804,096
    //   W2  : NPAD* 64*2 =  6,402,048
    //   t2  : NPAD*   4  =    200,064     (total ~30.7 MiB)
    char* ws = (char*)d_ws;
    bf16_t* Thi = (bf16_t*)ws;
    bf16_t* Tlo = (bf16_t*)(ws + (size_t)NPAD * DIM * 2);
    bf16_t* W2  = (bf16_t*)(ws + (size_t)NPAD * DIM * 4);
    float*  t2  = (float*)(ws + (size_t)NPAD * DIM * 4 + (size_t)NPAD * OUTD * 2);

    zero_out<<<(NFEAT * OUTD + 255) / 256, 256, 0, stream>>>(out, NFEAT * OUTD);
    prep_train<<<(NPAD + 7) / 8, 256, 0, stream>>>(train, Thi, Tlo, t2);
    prep_weights<<<(NPAD * 16 + 255) / 256, 256, 0, stream>>>(w, W2);

    dim3 grid(NFEAT / 16, SPLIT);
    rbf_fused<<<grid, 32, 0, stream>>>(feats, Thi, Tlo, t2, W2, out);
}
